// WindowBasedSelfAttention_55473797595513
// MI455X (gfx1250) — matrix operand order repaired and checked by reading, not compiled, not run on hardware
//
#include <hip/hip_runtime.h>

// ---------------------------------------------------------------------------
// WindowBasedSelfAttention for MI455X (gfx1250, wave32, WMMA)
// Heavy GEMMs run as bf16 v_wmma_f32_16x16x32_bf16 with f32 accumulation.
// Complex GEMM is mapped to one real GEMM via [[c,d],[-d,c]] weight expansion.
// Weights are pre-swizzled into WMMA fragment-major order (B bypasses LDS);
// A is LDS-staged in fragment-major order with double buffering.
// Wave tile = 64x16 (4 WMMAs sharing one B fragment): no duplicated B loads
// across the block, 4x the compute per streamed B byte.
// ---------------------------------------------------------------------------

typedef __attribute__((ext_vector_type(16))) __bf16 v16bf;
typedef __attribute__((ext_vector_type(8)))  float  v8f;

#define CDIV(a, b) (((a) + (b) - 1) / (b))

__device__ __forceinline__ unsigned short f2bf(float f) {
    union { float f; unsigned u; } x; x.f = f;
    unsigned u = x.u;
    u += 0x7fffu + ((u >> 16) & 1u);   // round-to-nearest-even
    return (unsigned short)(u >> 16);
}

// ---------------------------------------------------------------------------
// 64x64 FFT machinery (radix-2 DIT in LDS). sign=-1 forward, +1 inverse.
// ---------------------------------------------------------------------------
__device__ __forceinline__ void fft64_rows(float (*re)[66], float (*im)[66], float sign) {
    int tid = threadIdx.x;
    float tr[16], ti[16];
#pragma unroll
    for (int j = 0; j < 16; j++) {
        int e = tid + j * 256; int r = e >> 6, c = e & 63;
        int rc = __brev((unsigned)c) >> 26;
        tr[j] = re[r][rc]; ti[j] = im[r][rc];
    }
    __syncthreads();
#pragma unroll
    for (int j = 0; j < 16; j++) {
        int e = tid + j * 256; int r = e >> 6, c = e & 63;
        re[r][c] = tr[j]; im[r][c] = ti[j];
    }
    __syncthreads();
    for (int s = 1; s <= 6; s++) {
        int h = 1 << (s - 1);
        for (int t = tid; t < 2048; t += 256) {
            int line = t >> 5, k = t & 31;
            int pos = k & (h - 1), grp = k >> (s - 1);
            int i0 = (grp << s) + pos, i1 = i0 + h;
            float ang = sign * 6.283185307f * (float)pos / (float)(1 << s);
            float sw, cw; __sincosf(ang, &sw, &cw);
            float ur = re[line][i0], ui = im[line][i0];
            float vr = re[line][i1], vi = im[line][i1];
            float wr = vr * cw - vi * sw, wi = vr * sw + vi * cw;
            re[line][i0] = ur + wr; im[line][i0] = ui + wi;
            re[line][i1] = ur - wr; im[line][i1] = ui - wi;
        }
        __syncthreads();
    }
}

__device__ __forceinline__ void fft64_cols(float (*re)[66], float (*im)[66], float sign) {
    int tid = threadIdx.x;
    float tr[16], ti[16];
#pragma unroll
    for (int j = 0; j < 16; j++) {
        int e = tid + j * 256; int col = e >> 6, r = e & 63;
        int rr = __brev((unsigned)r) >> 26;
        tr[j] = re[rr][col]; ti[j] = im[rr][col];
    }
    __syncthreads();
#pragma unroll
    for (int j = 0; j < 16; j++) {
        int e = tid + j * 256; int col = e >> 6, r = e & 63;
        re[r][col] = tr[j]; im[r][col] = ti[j];
    }
    __syncthreads();
    for (int s = 1; s <= 6; s++) {
        int h = 1 << (s - 1);
        for (int t = tid; t < 2048; t += 256) {
            int line = t >> 5, k = t & 31;
            int pos = k & (h - 1), grp = k >> (s - 1);
            int i0 = (grp << s) + pos, i1 = i0 + h;
            float ang = sign * 6.283185307f * (float)pos / (float)(1 << s);
            float sw, cw; __sincosf(ang, &sw, &cw);
            float ur = re[i0][line], ui = im[i0][line];
            float vr = re[i1][line], vi = im[i1][line];
            float wr = vr * cw - vi * sw, wi = vr * sw + vi * cw;
            re[i0][line] = ur + wr; im[i0][line] = ui + wi;
            re[i1][line] = ur - wr; im[i1][line] = ui - wi;
        }
        __syncthreads();
    }
}

// Window extraction + forward FFT2 (ortho) + fftshift. One block = (b,c,win).
__global__ __launch_bounds__(256) void win_fft_kernel(const float* __restrict__ x,
                                                      float* __restrict__ four) {
    int bid = blockIdx.x;                 // (b*3+c)*225 + win
    int win = bid % 225; int bc = bid / 225;
    int wi = win / 15, wj = win % 15;
    __shared__ float re[64][66], im[64][66];
    int tid = threadIdx.x;
    const float* src = x + ((size_t)bc * 512 + (size_t)wi * 32) * 512 + wj * 32;
    for (int e = tid; e < 4096; e += 256) {
        int r = e >> 6, c = e & 63;
        re[r][c] = src[(size_t)r * 512 + c];
        im[r][c] = 0.f;
    }
    __syncthreads();
    fft64_rows(re, im, -1.f);
    fft64_cols(re, im, -1.f);
    float* dst = four + (size_t)bid * 8192;
    for (int e = tid; e < 4096; e += 256) {
        int u = e >> 6, v = e & 63;
        int su = (u + 32) & 63, sv = (v + 32) & 63;    // fftshift
        dst[(size_t)(su * 64 + sv) * 2]     = re[u][v] * (1.f / 64.f);
        dst[(size_t)(su * 64 + sv) * 2 + 1] = im[u][v] * (1.f / 64.f);
    }
}

// Per-element argmax over channels -> bf16 interleaved tokens (M=900 x 8192).
__global__ __launch_bounds__(256) void fold_kernel(const float* __restrict__ four,
                                                   unsigned short* __restrict__ toks) {
    size_t e = (size_t)blockIdx.x * 256 + threadIdx.x;
    if (e >= (size_t)4 * 225 * 4096) return;
    int f = (int)(e & 4095); size_t bw = e >> 12;
    int win = (int)(bw % 225); int b = (int)(bw / 225);
    float br = 0.f, bi = 0.f, best = -1.f;
#pragma unroll
    for (int c = 0; c < 3; c++) {
        size_t idx = (((size_t)(b * 3 + c) * 225 + win) * 4096 + f) * 2;
        float r = four[idx], i = four[idx + 1];
        float m = r * r + i * i;
        if (m > best) { best = m; br = r; bi = i; }
    }
    toks[e * 2]     = f2bf(br);
    toks[e * 2 + 1] = f2bf(bi);
}

// ---------------------------------------------------------------------------
// Fragment-major B layout helper.
// B[row][col] (row = k index of 2K, col = n index of 2N) lives at
//   ((kblk * (N2/16) + cblk) * 32 + lane) * 16 + i
// where kblk=row/32, half=(row%32)/16, i=row%16, cblk=col/16, l15=col%16,
// lane=half*16+l15 — contiguous k order (k = 16*half + i), same as A, so a lane
// fetches its whole fragment as two b128 loads.
// ---------------------------------------------------------------------------
__device__ __forceinline__ size_t fragidx(int row, int col, int nblk16) {
    int kblk = row >> 5, kk = row & 31;
    int half = kk >> 4, i = kk & 15;
    int cblk = col >> 4, l15 = col & 15;
    int lane = half * 16 + l15;
    return ((size_t)(kblk * nblk16 + cblk) * 32 + lane) * 16 + i;
}

// Expand complex weight W[N x K] (+bias[N]) into fragment-major real bf16
// B[2K x 2N] + f32 bias[2N].
__global__ __launch_bounds__(256) void expand_w(const float* __restrict__ w,
                                                const float* __restrict__ b,
                                                unsigned short* __restrict__ Bex,
                                                float* __restrict__ biasex,
                                                int N, int K) {
    size_t i = (size_t)blockIdx.x * 256 + threadIdx.x;
    if (i < (size_t)N * K) {
        int n = (int)(i / K), k = (int)(i % K);
        float wr = w[2 * i], wi = w[2 * i + 1];
        int nb = (2 * N) >> 4;
        Bex[fragidx(2 * k,     2 * n,     nb)] = f2bf(wr);
        Bex[fragidx(2 * k,     2 * n + 1, nb)] = f2bf(wi);
        Bex[fragidx(2 * k + 1, 2 * n,     nb)] = f2bf(-wi);
        Bex[fragidx(2 * k + 1, 2 * n + 1, nb)] = f2bf(wr);
    }
    if (i < (size_t)N) { biasex[2 * i] = b[2 * i]; biasex[2 * i + 1] = b[2 * i + 1]; }
}

// ---------------------------------------------------------------------------
// bf16 WMMA GEMM: C[MxN] = A[MxK] * Bfrag[KxN] + bias[N].
// 256 threads = 8 waves. Block tile 64x128; each wave owns one 16-col block
// (cblk = wave id) across all 64 rows: per k-step it loads ONE B fragment
// (2 coalesced global_load_b128, unique per wave -> zero duplication) and the
// four LDS-resident A fragments (8 ds_load_b128), issuing 4 WMMAs.
// A: LDS fragment-major, double-buffered (1 b128 global + 1 b128 ds-store per
//    thread per k-step). Next k-step of B is prefetched.
// ---------------------------------------------------------------------------
union FragU { uint4 q[2]; v16bf v; };

__global__ __launch_bounds__(256) void gemm_bf16_wmma(const unsigned short* __restrict__ A,
                                                      const unsigned short* __restrict__ Bf,
                                                      const float* __restrict__ bias,
                                                      float* __restrict__ C,
                                                      int M, int K, int N) {
    __shared__ uint4 sA[2][4][32][2];          // [buf][wm][lane][chunk] = 8 KB
    int tid = threadIdx.x, lane = tid & 31, wave = tid >> 5;
    int bm = blockIdx.y * 64, bn = blockIdx.x * 128;
    int half = lane >> 4, l15 = lane & 15;
    v8f acc[4] = {{}, {}, {}, {}};

    // Per-thread A staging assignment: (row, half, chunk).
    // Lane fragment = A[row][k0 + half*16 + chunk*8 .. +8) for chunk 0/1.
    int srow = tid >> 2;
    int shalf = (tid >> 1) & 1, schunk = tid & 1;
    int sgr = bm + srow; if (sgr >= M) sgr = M - 1;
    const unsigned short* aRow = A + (size_t)sgr * K + shalf * 16 + schunk * 8;
    int sdwm = srow >> 4, sdlane = shalf * 16 + (srow & 15);

    const uint4* Bq = (const uint4*)Bf;
    int nblk = N >> 4;
    int cb = (bn >> 4) + wave;                 // this wave's unique 16-col block
    size_t bstep = (size_t)nblk * 64;          // uint4 stride between k-blocks
    int nk = K >> 5;

    sA[0][sdwm][sdlane][schunk] = *(const uint4*)aRow;   // prologue stage

    for (int kb = 0; kb < nk; kb++) {
        int buf = kb & 1;
        __syncthreads();
        if (kb + 1 < nk) {
            sA[buf ^ 1][sdwm][sdlane][schunk] = *(const uint4*)(aRow + (size_t)(kb + 1) * 32);
        }
        FragU fb, fa0, fa1, fa2, fa3;
        size_t bb = ((size_t)kb * nblk + cb) * 64 + lane * 2;
        fb.q[0] = Bq[bb];
        fb.q[1] = Bq[bb + 1];
        fa0.q[0] = sA[buf][0][lane][0];  fa0.q[1] = sA[buf][0][lane][1];
        fa1.q[0] = sA[buf][1][lane][0];  fa1.q[1] = sA[buf][1][lane][1];
        fa2.q[0] = sA[buf][2][lane][0];  fa2.q[1] = sA[buf][2][lane][1];
        fa3.q[0] = sA[buf][3][lane][0];  fa3.q[1] = sA[buf][3][lane][1];
        if (kb + 1 < nk) {                     // prefetch next k-block of B
            __builtin_prefetch(Bq + bb + bstep, 0, 1);
        }
        acc[0] = __builtin_amdgcn_wmma_f32_16x16x32_bf16(false, fa0.v, false, fb.v, (short)0, acc[0], false, false);
        acc[1] = __builtin_amdgcn_wmma_f32_16x16x32_bf16(false, fa1.v, false, fb.v, (short)0, acc[1], false, false);
        acc[2] = __builtin_amdgcn_wmma_f32_16x16x32_bf16(false, fa2.v, false, fb.v, (short)0, acc[2], false, false);
        acc[3] = __builtin_amdgcn_wmma_f32_16x16x32_bf16(false, fa3.v, false, fb.v, (short)0, acc[3], false, false);
    }

    int col0 = bn + wave * 16 + l15;
    float bs = bias[col0];
#pragma unroll
    for (int m = 0; m < 4; m++) {
#pragma unroll
        for (int r = 0; r < 8; r++) {
            int row = bm + m * 16 + r + 8 * half;   // C layout: VGPR r -> M=r (lo) / r+8 (hi)
            if (row < M) C[(size_t)row * N + col0] = acc[m][r] + bs;
        }
    }
}

// ---------------------------------------------------------------------------
// Complex LayerNorm helpers
// ---------------------------------------------------------------------------
__device__ __forceinline__ float blockReduceSum(float v, float* red) {
    int tid = threadIdx.x;
    __syncthreads();
    red[tid] = v; __syncthreads();
    for (int s = 128; s > 0; s >>= 1) {
        if (tid < s) red[tid] += red[tid + s];
        __syncthreads();
    }
    return red[0];
}

// complex LN + leaky_relu(re)+i*leaky_relu(im) -> bf16 (feeds q/k/v GEMMs)
__global__ __launch_bounds__(256) void cln_lrelu_bf16(const float* __restrict__ Y,
                                                      const float* __restrict__ g,
                                                      const float* __restrict__ b,
                                                      unsigned short* __restrict__ out,
                                                      int d) {
    int row = blockIdx.x, tid = threadIdx.x;
    const float* y = Y + (size_t)row * 2 * d;
    __shared__ float red[256];
    __shared__ float mr_s, mi_s, inv_s;
    float sr = 0.f, si = 0.f;
    for (int i = tid; i < d; i += 256) { sr += y[2 * i]; si += y[2 * i + 1]; }
    sr = blockReduceSum(sr, red);
    si = blockReduceSum(si, red);
    if (tid == 0) { mr_s = sr / d; mi_s = si / d; }
    __syncthreads();
    float mr = mr_s, mi = mi_s, var = 0.f;
    for (int i = tid; i < d; i += 256) {
        float dr = y[2 * i] - mr, di = y[2 * i + 1] - mi;
        var += dr * dr + di * di;
    }
    var = blockReduceSum(var, red);
    if (tid == 0) inv_s = rsqrtf(var / d + 1e-5f);
    __syncthreads();
    float inv = inv_s;
    for (int i = tid; i < d; i += 256) {
        float xr = (y[2 * i] - mr) * inv, xi = (y[2 * i + 1] - mi) * inv;
        float gr = g[2 * i], gi = g[2 * i + 1];
        float zr = gr * xr - gi * xi + b[2 * i];
        float zi = gr * xi + gi * xr + b[2 * i + 1];
        zr = zr > 0.f ? zr : 0.01f * zr;
        zi = zi > 0.f ? zi : 0.01f * zi;
        out[(size_t)row * 2 * d + 2 * i]     = f2bf(zr);
        out[(size_t)row * 2 * d + 2 * i + 1] = f2bf(zi);
    }
}

// complex LN + complex sigmoid -> filt (f32 interleaved)
__global__ __launch_bounds__(256) void cln_sigmoid(const float* __restrict__ Y,
                                                   const float* __restrict__ g,
                                                   const float* __restrict__ b,
                                                   float* __restrict__ filt,
                                                   int d) {
    int row = blockIdx.x, tid = threadIdx.x;
    const float* y = Y + (size_t)row * 2 * d;
    __shared__ float red[256];
    __shared__ float mr_s, mi_s, inv_s;
    float sr = 0.f, si = 0.f;
    for (int i = tid; i < d; i += 256) { sr += y[2 * i]; si += y[2 * i + 1]; }
    sr = blockReduceSum(sr, red);
    si = blockReduceSum(si, red);
    if (tid == 0) { mr_s = sr / d; mi_s = si / d; }
    __syncthreads();
    float mr = mr_s, mi = mi_s, var = 0.f;
    for (int i = tid; i < d; i += 256) {
        float dr = y[2 * i] - mr, di = y[2 * i + 1] - mi;
        var += dr * dr + di * di;
    }
    var = blockReduceSum(var, red);
    if (tid == 0) inv_s = rsqrtf(var / d + 1e-5f);
    __syncthreads();
    float inv = inv_s;
    for (int i = tid; i < d; i += 256) {
        float xr = (y[2 * i] - mr) * inv, xi = (y[2 * i + 1] - mi) * inv;
        float gr = g[2 * i], gi = g[2 * i + 1];
        float zr = gr * xr - gi * xi + b[2 * i];
        float zi = gr * xi + gi * xr + b[2 * i + 1];
        // 1/(1+exp(-z)) for complex z
        float er = expf(-zr);
        float sb, cb; __sincosf(zi, &sb, &cb);
        float wr = 1.f + er * cb, wi = -er * sb;
        float den = wr * wr + wi * wi;
        filt[(size_t)row * 2 * d + 2 * i]     = wr / den;
        filt[(size_t)row * 2 * d + 2 * i + 1] = -wi / den;
    }
}

// scores = (q . k)/sqrt(d) complex; softmax over |scores|. One block = (b,l).
__global__ __launch_bounds__(256) void attn_scores(const float* __restrict__ Q,
                                                   const float* __restrict__ Km,
                                                   float* __restrict__ attn) {
    int row = blockIdx.x; int b = row / 225, l = row % 225;
    int tid = threadIdx.x;
    __shared__ float s[256];
    __shared__ float red[256];
    float mag = -1e30f;
    if (tid < 225) {
        const float* q = Q + ((size_t)b * 225 + l) * 4096;
        const float* k = Km + ((size_t)b * 225 + tid) * 4096;
        float sr = 0.f, si = 0.f;
        for (int i = 0; i < 2048; i++) {
            float qr = q[2 * i], qi = q[2 * i + 1];
            float kr = k[2 * i], ki = k[2 * i + 1];
            sr += qr * kr - qi * ki;
            si += qr * ki + qi * kr;
        }
        const float scale = 0.02209708691f;   // 1/sqrt(2048)
        sr *= scale; si *= scale;
        mag = sqrtf(sr * sr + si * si);
    }
    s[tid] = mag;
    red[tid] = mag; __syncthreads();
    for (int st = 128; st > 0; st >>= 1) {
        if (tid < st) red[tid] = fmaxf(red[tid], red[tid + st]);
        __syncthreads();
    }
    float mx = red[0]; __syncthreads();
    float e = (tid < 225) ? __expf(s[tid] - mx) : 0.f;
    red[tid] = e; __syncthreads();
    for (int st = 128; st > 0; st >>= 1) {
        if (tid < st) red[tid] += red[tid + st];
        __syncthreads();
    }
    float sum = red[0];
    if (tid < 225) attn[(size_t)row * 225 + tid] = e / sum;
}

// y = attn @ v (attn real). One block = (b,l).
__global__ __launch_bounds__(256) void apply_attn(const float* __restrict__ attn,
                                                  const float* __restrict__ V,
                                                  float* __restrict__ Y,
                                                  unsigned short* __restrict__ Ybf) {
    int row = blockIdx.x; int b = row / 225;
    __shared__ float sa[225];
    for (int i = threadIdx.x; i < 225; i += 256) sa[i] = attn[(size_t)row * 225 + i];
    __syncthreads();
    for (int j = threadIdx.x; j < 4096; j += 256) {
        float acc = 0.f;
        const float* v = V + ((size_t)b * 225) * 4096 + j;
        for (int m = 0; m < 225; m++) acc += sa[m] * v[(size_t)m * 4096];
        Y[(size_t)row * 4096 + j] = acc;
        Ybf[(size_t)row * 4096 + j] = f2bf(acc);
    }
}

// o = real(ifft2(ifftshift(four*filt), ortho)). One block = (b,c,win).
__global__ __launch_bounds__(256) void ifft_kernel(const float* __restrict__ four,
                                                   const float* __restrict__ filt,
                                                   float* __restrict__ o) {
    int bid = blockIdx.x;                 // (b*3+c)*225 + win
    int win = bid % 225; int bc = bid / 225; int b = bc / 3;
    __shared__ float re[64][66], im[64][66];
    int tid = threadIdx.x;
    const float* F  = four + (size_t)bid * 8192;
    const float* Ft = filt + ((size_t)b * 225 + win) * 8192;
    for (int e = tid; e < 4096; e += 256) {
        int u = e >> 6, v = e & 63;
        int su = (u + 32) & 63, sv = (v + 32) & 63;   // ifftshift (even N)
        int si2 = (su * 64 + sv) * 2;
        float ar = F[si2], ai = F[si2 + 1];
        float fr = Ft[si2], fi = Ft[si2 + 1];
        re[u][v] = ar * fr - ai * fi;
        im[u][v] = ar * fi + ai * fr;
    }
    __syncthreads();
    fft64_rows(re, im, 1.f);
    fft64_cols(re, im, 1.f);
    for (int e = tid; e < 4096; e += 256) {
        int u = e >> 6, v = e & 63;
        o[(size_t)bid * 4096 + e] = re[u][v] * (1.f / 64.f);
    }
}

// Overlap-add merge of even/odd windows -> out (4,3,512,512)
__global__ __launch_bounds__(256) void merge_out(const float* __restrict__ o,
                                                 float* __restrict__ out) {
    size_t e = (size_t)blockIdx.x * 256 + threadIdx.x;
    if (e >= (size_t)4 * 3 * 512 * 512) return;
    int col = (int)(e & 511); size_t t = e >> 9;
    int rowp = (int)(t & 511); t >>= 9;
    int bc = (int)t;                        // b*3 + ch
    int gi = rowp >> 6, u = rowp & 63, gj = col >> 6, v = col & 63;
    const float* base = o + (size_t)bc * 225 * 4096;
    auto O = [&](int wi, int wj, int uu, int vv) -> float {
        return base[(size_t)(wi * 15 + wj) * 4096 + uu * 64 + vv];
    };
    float val = O(2 * gi, 2 * gj, u, v);
    if (u >= 32 && v >= 32) { if (gi < 7 && gj < 7)  val = 0.5f * val + 0.5f * O(2 * gi + 1, 2 * gj + 1, u - 32, v - 32); }
    else if (u >= 32)       { if (gi < 7 && gj >= 1) val = 0.5f * val + 0.5f * O(2 * gi + 1, 2 * gj - 1, u - 32, v + 32); }
    else if (v >= 32)       { if (gi >= 1 && gj < 7) val = 0.5f * val + 0.5f * O(2 * gi - 1, 2 * gj + 1, u + 32, v - 32); }
    else                    { if (gi >= 1 && gj >= 1) val = 0.5f * val + 0.5f * O(2 * gi - 1, 2 * gj - 1, u + 32, v + 32); }
    out[e] = val;
}

// merge(filt) -> (4,1,960,960) complex, written interleaved
__global__ __launch_bounds__(256) void merge_filt(const float* __restrict__ filt,
                                                  float* __restrict__ out) {
    size_t e = (size_t)blockIdx.x * 256 + threadIdx.x;
    if (e >= (size_t)4 * 960 * 960) return;
    int col = (int)(e % 960); size_t t = e / 960;
    int rowp = (int)(t % 960); int b = (int)(t / 960);
    int i = rowp >> 6, u = rowp & 63, j = col >> 6, v = col & 63;
    size_t src = ((size_t)b * 225 + (i * 15 + j)) * 8192 + (size_t)(u * 64 + v) * 2;
    out[2 * e]     = filt[src];
    out[2 * e + 1] = filt[src + 1];
}

// ---------------------------------------------------------------------------
extern "C" void kernel_launch(void* const* d_in, const int* in_sizes, int n_in,
                              void* d_out, int out_size, void* d_ws, size_t ws_size,
                              hipStream_t stream) {
    const float* x     = (const float*)d_in[0];
    const float* l1_w  = (const float*)d_in[1];
    const float* l1_b  = (const float*)d_in[2];
    const float* ln1_g = (const float*)d_in[3];
    const float* ln1_b = (const float*)d_in[4];
    const float* q_w   = (const float*)d_in[5];
    const float* q_b   = (const float*)d_in[6];
    const float* k_w   = (const float*)d_in[7];
    const float* k_b   = (const float*)d_in[8];
    const float* v_w   = (const float*)d_in[9];
    const float* v_b   = (const float*)d_in[10];
    const float* l2_w  = (const float*)d_in[11];
    const float* l2_b  = (const float*)d_in[12];
    const float* ln2_g = (const float*)d_in[13];
    const float* ln2_b = (const float*)d_in[14];
    float* out = (float*)d_out;

    char* wp = (char*)d_ws;
    auto carve = [&](size_t bytes) -> void* {
        void* r = (void*)wp;
        wp += (bytes + 255) & ~(size_t)255;
        return r;
    };

    const int M = 900;                       // 4 batches x 225 windows
    float*          four    = (float*)carve((size_t)2700 * 8192 * 4);        // B,C,win spectra
    unsigned short* toksbf  = (unsigned short*)carve((size_t)M * 8192 * 2);
    unsigned short* WexL1   = (unsigned short*)carve((size_t)8192 * 4096 * 2);
    float*          biasL1  = (float*)carve(4096 * 4);
    unsigned short* WexQ    = (unsigned short*)carve((size_t)4096 * 4096 * 2);
    float*          biasQ   = (float*)carve(4096 * 4);
    unsigned short* WexK    = (unsigned short*)carve((size_t)4096 * 4096 * 2);
    float*          biasK   = (float*)carve(4096 * 4);
    unsigned short* WexV    = (unsigned short*)carve((size_t)4096 * 4096 * 2);
    float*          biasV   = (float*)carve(4096 * 4);
    unsigned short* WexL2   = (unsigned short*)carve((size_t)4096 * 8192 * 2);
    float*          biasL2  = (float*)carve(8192 * 4);
    float*          y1      = (float*)carve((size_t)M * 4096 * 4);
    unsigned short* ybf     = (unsigned short*)carve((size_t)M * 4096 * 2);
    float*          qf      = (float*)carve((size_t)M * 4096 * 4);
    float*          kf      = (float*)carve((size_t)M * 4096 * 4);
    float*          vf      = (float*)carve((size_t)M * 4096 * 4);
    float*          attn    = (float*)carve((size_t)4 * 225 * 225 * 4);
    float*          y2      = (float*)carve((size_t)M * 4096 * 4);
    unsigned short* y2bf    = (unsigned short*)carve((size_t)M * 4096 * 2);
    float*          l2out   = (float*)carve((size_t)M * 8192 * 4);
    float*          filt    = (float*)carve((size_t)M * 8192 * 4);
    float*          o       = (float*)carve((size_t)2700 * 4096 * 4);

    // 1) windows -> FFT2 (shifted) spectra
    win_fft_kernel<<<2700, 256, 0, stream>>>(x, four);
    // 2) channel-argmax fold -> bf16 tokens
    fold_kernel<<<CDIV((size_t)4 * 225 * 4096, 256), 256, 0, stream>>>(four, toksbf);
    // 3) weight expansion (complex -> fragment-major real bf16)
    expand_w<<<CDIV((size_t)2048 * 4096, 256), 256, 0, stream>>>(l1_w, l1_b, WexL1, biasL1, 2048, 4096);
    expand_w<<<CDIV((size_t)2048 * 2048, 256), 256, 0, stream>>>(q_w, q_b, WexQ, biasQ, 2048, 2048);
    expand_w<<<CDIV((size_t)2048 * 2048, 256), 256, 0, stream>>>(k_w, k_b, WexK, biasK, 2048, 2048);
    expand_w<<<CDIV((size_t)2048 * 2048, 256), 256, 0, stream>>>(v_w, v_b, WexV, biasV, 2048, 2048);
    expand_w<<<CDIV((size_t)4096 * 2048, 256), 256, 0, stream>>>(l2_w, l2_b, WexL2, biasL2, 4096, 2048);
    // 4) l1 GEMM (900 x 8192 x 4096) via WMMA
    gemm_bf16_wmma<<<dim3(4096 / 128, CDIV(M, 64)), 256, 0, stream>>>(toksbf, WexL1, biasL1, y1, M, 8192, 4096);
    // 5) complex LN + leaky relu -> bf16
    cln_lrelu_bf16<<<M, 256, 0, stream>>>(y1, ln1_g, ln1_b, ybf, 2048);
    // 6) q/k/v GEMMs (900 x 4096 x 4096)
    gemm_bf16_wmma<<<dim3(4096 / 128, CDIV(M, 64)), 256, 0, stream>>>(ybf, WexQ, biasQ, qf, M, 4096, 4096);
    gemm_bf16_wmma<<<dim3(4096 / 128, CDIV(M, 64)), 256, 0, stream>>>(ybf, WexK, biasK, kf, M, 4096, 4096);
    gemm_bf16_wmma<<<dim3(4096 / 128, CDIV(M, 64)), 256, 0, stream>>>(ybf, WexV, biasV, vf, M, 4096, 4096);
    // 7) attention
    attn_scores<<<M, 256, 0, stream>>>(qf, kf, attn);
    apply_attn<<<M, 256, 0, stream>>>(attn, vf, y2, y2bf);
    // 8) l2 GEMM (900 x 4096 x 8192)
    gemm_bf16_wmma<<<dim3(8192 / 128, CDIV(M, 64)), 256, 0, stream>>>(y2bf, WexL2, biasL2, l2out, M, 4096, 8192);
    // 9) complex LN + complex sigmoid -> filter
    cln_sigmoid<<<M, 256, 0, stream>>>(l2out, ln2_g, ln2_b, filt, 4096);
    // 10) filter * spectrum -> inverse FFT2 -> real
    ifft_kernel<<<2700, 256, 0, stream>>>(four, filt, o);
    // 11) overlap-add merge + filt merge into d_out
    merge_out<<<CDIV((size_t)4 * 3 * 512 * 512, 256), 256, 0, stream>>>(o, out);
    merge_filt<<<CDIV((size_t)4 * 960 * 960, 256), 256, 0, stream>>>(filt, out + (size_t)4 * 3 * 512 * 512);
}
